// GraphNetPairwise_13219909337154
// MI455X (gfx1250) — matrix operand order repaired and checked by reading, not compiled, not run on hardware
//
#include <hip/hip_runtime.h>

// ---------------------------------------------------------------------------
// GraphNet on MI455X (gfx1250): f16 WMMA GEMMs with fused gather/concat,
// f32 atomics for scatter / segment sums, weights pre-packed into the
// WMMA B-fragment layout (ISA 7.12.2).
// Wave tile 16(M) x 128(N): one A fragment feeds 8 v_wmma_f32_16x16x32_f16.
// ---------------------------------------------------------------------------

#define N_ATOMS 65536
#define N_BONDS 98304
#define N_MOLS  4096
#define D_IN    128
#define DM      256   // model dim

typedef _Float16 v16h __attribute__((ext_vector_type(16)));
typedef float    v8f  __attribute__((ext_vector_type(8)));

struct GemmSrcs {
    const void* ptr[5];   // per-256(or 128)-wide concat chunk
    const int*  idx[5];   // row gather indices (nullptr = identity)
    int         isF32[5]; // 0 = _Float16 source, 1 = float source
};

// ---------------------------------------------------------------------------
// Pack a [K, 256] f32 weight into WMMA B-fragment layout:
//   Wpk[(((kt*16 + nt)*32 + lane)*16) + j]  with
//   n = nt*16 + (lane&15),  k = kt*32 + ((lane>>4)<<3) + (j<8 ? j : j+8)
// so the GEMM loads one aligned v16h (32 B) per lane per 16-col subtile.
// ---------------------------------------------------------------------------
__global__ __launch_bounds__(256) void pack_weight_kernel(
    const float* __restrict__ W, _Float16* __restrict__ Wpk)
{
    int tid  = blockIdx.x * 256 + threadIdx.x;
    int j    = tid & 15;
    int lane = (tid >> 4) & 31;
    int nt   = (tid >> 9) & 15;
    int kt   = tid >> 13;
    int n = nt * 16 + (lane & 15);
    int k = kt * 32 + ((lane >> 4) << 3) + (j < 8 ? j : j + 8);
    Wpk[tid] = (_Float16)W[(size_t)k * DM + n];
}

// One K-step: 8 WMMAs against one A fragment, B loaded in groups of 4 so the
// scheduler can overlap loads with XDL execution (decreasing loadcnt waits).
__device__ __forceinline__ void gemm_step(
    const v16h& a, const _Float16* __restrict__ Wpk,
    size_t ktG, int ntBase, int lane, v8f acc[8])
{
    const _Float16* wb = Wpk + (((ktG * 16 + ntBase) * 32 + lane) << 4);
    __builtin_prefetch(wb + ((size_t)16 << 9), 0, 1);  // next kt's B panel
    v16h b[4];
#pragma unroll
    for (int sub = 0; sub < 4; ++sub) b[sub] = *(const v16h*)(wb + (sub << 9));
#pragma unroll
    for (int sub = 0; sub < 4; ++sub)
        acc[sub] = __builtin_amdgcn_wmma_f32_16x16x32_f16(
            false, a, false, b[sub], (short)0, acc[sub], false, false);
#pragma unroll
    for (int sub = 0; sub < 4; ++sub) b[sub] = *(const v16h*)(wb + ((sub + 4) << 9));
#pragma unroll
    for (int sub = 0; sub < 4; ++sub)
        acc[sub + 4] = __builtin_amdgcn_wmma_f32_16x16x32_f16(
            false, a, false, b[sub], (short)0, acc[sub + 4], false, false);
}

// ---------------------------------------------------------------------------
// Fused concat-gather WMMA GEMM + tanh:
//   out = tanh( concat_s( src_s[idx_s[row]] ) @ W )   with W pre-packed.
// Block: 256 threads (8 waves), tile 128(M) x 128(N); wave tile 16 x 128.
// ---------------------------------------------------------------------------
__global__ __launch_bounds__(256) void wmma_gemm_tanh_kernel(
    GemmSrcs srcs, int S, int cwShift /* log2 chunk width: 7 or 8 */,
    const _Float16* __restrict__ Wpk, int M,
    _Float16* __restrict__ outH, float* __restrict__ outF)
{
    const int lane   = threadIdx.x & 31;
    const int wave   = threadIdx.x >> 5;
    const int laneHi = lane >> 4;
    const int mBase  = blockIdx.x * 128 + wave * 16;
    const int nBase  = blockIdx.y * 128;
    const int ntBase = nBase >> 4;
    const int row    = mBase + (lane & 15);
    const int CW     = 1 << cwShift;

    v8f acc[8] = {};
    size_t ktGlobal = 0;

    for (int s = 0; s < S; ++s) {
        const int*  ix    = srcs.idx[s];
        const int   r     = ix ? ix[row] : row;
        const char* base  = (const char*)srcs.ptr[s];
        const int   steps = CW >> 5;

        if (!srcs.isF32[s]) {
            // f16 source: A layout groups k=0..7 and k=16..23 contiguously
            // per lane -> two 16B loads (compiler emits global_load_b128 x2).
            const _Float16* p0 = (const _Float16*)base + (size_t)r * CW + (laneHi << 3);
            for (int t = 0; t < steps; ++t, ++ktGlobal) {
                const _Float16* p = p0 + (t << 5);
                union { v16h v; _Float16 h[16]; } a;
#pragma unroll
                for (int q = 0; q < 8; ++q) {
                    const int kl = (q < 4) ? (q * 2) : (q * 2 + 8);
                    a.h[2 * q]     = p[kl];
                    a.h[2 * q + 1] = p[kl + 1];
                }
                gemm_step(a.v, Wpk, ktGlobal, ntBase, lane, acc);
            }
        } else {
            // f32 source (e_bar_v / e_bar / v_bar / raw atoms): load + convert.
            const float* p0 = (const float*)base + (size_t)r * CW + (laneHi << 3);
            for (int t = 0; t < steps; ++t, ++ktGlobal) {
                const float* p = p0 + (t << 5);
                union { v16h v; _Float16 h[16]; } a;
#pragma unroll
                for (int q = 0; q < 8; ++q) {
                    const int kl = (q < 4) ? (q * 2) : (q * 2 + 8);
                    float2 f = *(const float2*)(p + kl);
                    a.h[2 * q]     = (_Float16)f.x;
                    a.h[2 * q + 1] = (_Float16)f.y;
                }
                gemm_step(a.v, Wpk, ktGlobal, ntBase, lane, acc);
            }
        }
    }

    // ---- epilogue: tanh, store f16 (state) and/or f32 (final output) ----
    const int nCol = nBase + (lane & 15);
#pragma unroll
    for (int sub = 0; sub < 8; ++sub) {
#pragma unroll
        for (int rr = 0; rr < 8; ++rr) {
            const int m = mBase + rr + (laneHi << 3);
            const int n = nCol + sub * 16;
            const float v = tanhf(acc[sub][rr]);
            if (outH) outH[(size_t)m * DM + n] = (_Float16)v;
            if (outF) outF[(size_t)m * DM + n] = v;
        }
    }
}

// ---------------------------------------------------------------------------
// Small helper kernels
// ---------------------------------------------------------------------------
__global__ void zero_f32_kernel(float* __restrict__ p, int n)
{
    int i = blockIdx.x * 256 + threadIdx.x;
    if (i < n) p[i] = 0.0f;
}

// h_e0 = tanh(bond_orders[:,None] * W_fe[0,:])  (rank-1, no GEMM needed)
__global__ void init_he0_kernel(const float* __restrict__ bo,
                                const float* __restrict__ wfe,
                                _Float16* __restrict__ he0)
{
    int b = blockIdx.x, c = threadIdx.x;
    he0[(size_t)b * DM + c] = (_Float16)tanhf(bo[b] * wfe[c]);
}

// per-molecule atom sum + counts (for mol_mean)
__global__ void atom_stats_kernel(const float* __restrict__ atoms,
                                  const int* __restrict__ amol,
                                  float* __restrict__ mm, float* __restrict__ cnt)
{
    int a = blockIdx.x, c = threadIdx.x;  // blockDim = 128
    int mol = amol[a];
    atomicAdd(&mm[(size_t)mol * D_IN + c], atoms[(size_t)a * D_IN + c]);
    if (c == 0) atomicAdd(&cnt[mol], 1.0f);
}

__global__ void mean_div_kernel(float* __restrict__ mm, const float* __restrict__ cnt)
{
    int i = blockIdx.x * 256 + threadIdx.x;  // N_MOLS * D_IN
    mm[i] /= fmaxf(cnt[i >> 7], 1.0f);
}

// e_bar_v[src] += h_e ; e_bar_v[dst] += h_e ; e_bar[bond_mol] += h_e
__global__ void scatter_edges_kernel(const _Float16* __restrict__ he,
                                     const int* __restrict__ src,
                                     const int* __restrict__ dst,
                                     const int* __restrict__ bmol,
                                     float* __restrict__ ebv,
                                     float* __restrict__ eb)
{
    int b = blockIdx.x, c = threadIdx.x;  // blockDim = 256
    float v = (float)he[(size_t)b * DM + c];
    atomicAdd(&ebv[(size_t)src[b] * DM + c], v);
    atomicAdd(&ebv[(size_t)dst[b] * DM + c], v);
    atomicAdd(&eb[(size_t)bmol[b] * DM + c], v);
}

// v_bar[atom_mol] += h_v
__global__ void scatter_atoms_kernel(const _Float16* __restrict__ hv,
                                     const int* __restrict__ amol,
                                     float* __restrict__ vbar)
{
    int a = blockIdx.x, c = threadIdx.x;  // blockDim = 256
    atomicAdd(&vbar[(size_t)amol[a] * DM + c], (float)hv[(size_t)a * DM + c]);
}

// ---------------------------------------------------------------------------
// Host driver
// ---------------------------------------------------------------------------
static void launch_gemm(const GemmSrcs& s, int S, int cwShift, const _Float16* wpk,
                        int M, _Float16* outH, float* outF, hipStream_t st)
{
    dim3 grid(M / 128, 2);
    wmma_gemm_tanh_kernel<<<grid, 256, 0, st>>>(s, S, cwShift, wpk, M, outH, outF);
}

static void launch_zero(float* p, int n, hipStream_t st)
{
    zero_f32_kernel<<<(n + 255) / 256, 256, 0, st>>>(p, n);
}

extern "C" void kernel_launch(void* const* d_in, const int* in_sizes, int n_in,
                              void* d_out, int out_size, void* d_ws, size_t ws_size,
                              hipStream_t stream)
{
    (void)in_sizes; (void)n_in; (void)out_size; (void)ws_size;

    const float* atoms    = (const float*)d_in[0];
    const float* bond_ord = (const float*)d_in[1];
    const int*   bond_src = (const int*)d_in[2];
    const int*   bond_dst = (const int*)d_in[3];
    const int*   atom_mol = (const int*)d_in[4];
    const int*   bond_mol = (const int*)d_in[5];
    const float* W_fe = (const float*)d_in[6];
    const float* W_fv = (const float*)d_in[7];
    const float* W_fu = (const float*)d_in[8];
    const float* W_e  = (const float*)d_in[9];
    const float* W_v  = (const float*)d_in[10];
    const float* W_u  = (const float*)d_in[11];
    float* out = (float*)d_out;

    // ---- carve workspace (256B aligned slices) ----
    char* wp = (char*)d_ws;
    auto carve = [&](size_t bytes) -> void* {
        void* r = wp;
        wp += (bytes + 255) & ~(size_t)255;
        return r;
    };
    _Float16* wpk_e  = (_Float16*)carve((size_t)1280 * DM * 2);
    _Float16* wpk_v  = (_Float16*)carve((size_t)1024 * DM * 2);
    _Float16* wpk_u  = (_Float16*)carve((size_t)1024 * DM * 2);
    _Float16* wpk_fv = (_Float16*)carve((size_t)D_IN * DM * 2);
    _Float16* wpk_fu = (_Float16*)carve((size_t)D_IN * DM * 2);
    _Float16* he0 = (_Float16*)carve((size_t)N_BONDS * DM * 2);
    _Float16* heA = (_Float16*)carve((size_t)N_BONDS * DM * 2);
    _Float16* heB = (_Float16*)carve((size_t)N_BONDS * DM * 2);
    _Float16* hv0 = (_Float16*)carve((size_t)N_ATOMS * DM * 2);
    _Float16* hvA = (_Float16*)carve((size_t)N_ATOMS * DM * 2);
    _Float16* hvB = (_Float16*)carve((size_t)N_ATOMS * DM * 2);
    _Float16* hu0 = (_Float16*)carve((size_t)N_MOLS * DM * 2);
    _Float16* huA = (_Float16*)carve((size_t)N_MOLS * DM * 2);
    _Float16* huB = (_Float16*)carve((size_t)N_MOLS * DM * 2);
    float* ebv = (float*)carve((size_t)N_ATOMS * DM * 4);
    float* eb  = (float*)carve((size_t)N_MOLS * DM * 4);
    float* vb  = (float*)carve((size_t)N_MOLS * DM * 4);
    float* mm  = (float*)carve((size_t)N_MOLS * D_IN * 4);
    float* cnt = (float*)carve((size_t)N_MOLS * 4);

    // ---- pack weights into WMMA B-fragment layout ----
    pack_weight_kernel<<<(1280 * DM) / 256, 256, 0, stream>>>(W_e, wpk_e);
    pack_weight_kernel<<<(1024 * DM) / 256, 256, 0, stream>>>(W_v, wpk_v);
    pack_weight_kernel<<<(1024 * DM) / 256, 256, 0, stream>>>(W_u, wpk_u);
    pack_weight_kernel<<<(D_IN * DM) / 256, 256, 0, stream>>>(W_fv, wpk_fv);
    pack_weight_kernel<<<(D_IN * DM) / 256, 256, 0, stream>>>(W_fu, wpk_fu);

    // ---- initial embeddings ----
    init_he0_kernel<<<N_BONDS, DM, 0, stream>>>(bond_ord, W_fe, he0);

    launch_zero(mm, N_MOLS * D_IN, stream);
    launch_zero(cnt, N_MOLS, stream);
    atom_stats_kernel<<<N_ATOMS, D_IN, 0, stream>>>(atoms, atom_mol, mm, cnt);
    mean_div_kernel<<<(N_MOLS * D_IN) / 256, 256, 0, stream>>>(mm, cnt);

    {   // h_v0 = tanh(atoms @ W_fv)      M=65536, K=128
        GemmSrcs s = {};
        s.ptr[0] = atoms; s.idx[0] = nullptr; s.isF32[0] = 1;
        launch_gemm(s, 1, 7, wpk_fv, N_ATOMS, hv0, nullptr, stream);
    }
    {   // h_u0 = tanh(mol_mean @ W_fu)   M=4096, K=128
        GemmSrcs s = {};
        s.ptr[0] = mm; s.idx[0] = nullptr; s.isF32[0] = 1;
        launch_gemm(s, 1, 7, wpk_fu, N_MOLS, hu0, nullptr, stream);
    }

    // ---- message-passing iterations ----
    const _Float16* he_cur = he0;
    const _Float16* hv_cur = hv0;
    const _Float16* hu_cur = hu0;
    _Float16* heN[3] = {heA, heB, nullptr};  // last iter: f32 straight to d_out
    _Float16* hvN[2] = {hvA, hvB};
    _Float16* huN[2] = {huA, huB};

    for (int it = 0; it < 3; ++it) {
        const bool last = (it == 2);

        // h_e = tanh([h_e, h_e0, h_v[src], h_v[dst], h_u[bond_mol]] @ W_e)
        GemmSrcs se = {};
        se.ptr[0] = he_cur;  se.idx[0] = nullptr;  se.isF32[0] = 0;
        se.ptr[1] = he0;     se.idx[1] = nullptr;  se.isF32[1] = 0;
        se.ptr[2] = hv_cur;  se.idx[2] = bond_src; se.isF32[2] = 0;
        se.ptr[3] = hv_cur;  se.idx[3] = bond_dst; se.isF32[3] = 0;
        se.ptr[4] = hu_cur;  se.idx[4] = bond_mol; se.isF32[4] = 0;
        launch_gemm(se, 5, 8, wpk_e, N_BONDS, heN[it], last ? out : nullptr, stream);
        if (last) break;  // only h_e is returned; node/global updates dead

        // e_bar_v (both endpoints) and e_bar (per-mol edge sum)
        launch_zero(ebv, N_ATOMS * DM, stream);
        launch_zero(eb, N_MOLS * DM, stream);
        scatter_edges_kernel<<<N_BONDS, DM, 0, stream>>>(heN[it], bond_src, bond_dst,
                                                         bond_mol, ebv, eb);

        // h_v = tanh([h_v, h_v0, e_bar_v, h_u[atom_mol]] @ W_v)
        GemmSrcs sv = {};
        sv.ptr[0] = hv_cur; sv.idx[0] = nullptr;  sv.isF32[0] = 0;
        sv.ptr[1] = hv0;    sv.idx[1] = nullptr;  sv.isF32[1] = 0;
        sv.ptr[2] = ebv;    sv.idx[2] = nullptr;  sv.isF32[2] = 1;
        sv.ptr[3] = hu_cur; sv.idx[3] = atom_mol; sv.isF32[3] = 0;
        launch_gemm(sv, 4, 8, wpk_v, N_ATOMS, hvN[it], nullptr, stream);

        // v_bar (per-mol node sum)
        launch_zero(vb, N_MOLS * DM, stream);
        scatter_atoms_kernel<<<N_ATOMS, DM, 0, stream>>>(hvN[it], atom_mol, vb);

        // h_u = tanh([h_u, h_u0, e_bar, v_bar] @ W_u)
        GemmSrcs su = {};
        su.ptr[0] = hu_cur; su.idx[0] = nullptr; su.isF32[0] = 0;
        su.ptr[1] = hu0;    su.idx[1] = nullptr; su.isF32[1] = 0;
        su.ptr[2] = eb;     su.idx[2] = nullptr; su.isF32[2] = 1;
        su.ptr[3] = vb;     su.idx[3] = nullptr; su.isF32[3] = 1;
        launch_gemm(su, 4, 8, wpk_u, N_MOLS, huN[it], nullptr, stream);

        he_cur = heN[it];
        hv_cur = hvN[it];
        hu_cur = huN[it];
    }
}
